// Raindrop_12206297055894
// MI455X (gfx1250) — matrix operand order repaired and checked by reading, not compiled, not run on hardware
//
#include <hip/hip_runtime.h>

// ---------------- types for WMMA ----------------
typedef __attribute__((ext_vector_type(16))) __bf16 v16bf;
typedef __attribute__((ext_vector_type(8)))  float  v8f;

union FragAB { v16bf v; uint4 q[2]; };
union FragC  { v8f   v; float f[8]; };

// ---------------- problem constants ----------------
#define TT    215
#define TTP   224      // TT padded to K-multiple of 32
#define BB    256
#define VV    36
#define PED   16
#define DMOD  144
#define DMODP 160
#define DTR   720
#define DTRP  736
#define INCH  860
#define INCHP 864
#define NHIDD 512
#define HDIM  180
#define HDP   192
#define EE    1296
#define TBTOK 55040LL  // T*B
#define BH    1024     // B*NHEADS

// =====================================================================
// Tiled WMMA bf16 GEMM:  C = ACT( scale*(A @ B^T) + bias + res )
//   A: [M,K] bf16 (lda).  B: [N,K] bf16 (ldb), i.e. pre-transposed.
//   REQUIREMENTS (guaranteed by construction in this file):
//     K % 32 == 0, lda % 8 == 0, ldb % 8 == 0,
//     K-pad columns of B are zero (so A pad/junk never contributes).
//   block: 256 thr (8 waves). Block tile 128x64, wave tile 32x32.
//   Out-of-range rows/cols are staged from a clamped (valid) row and
//   discarded in the epilogue — staging is fully branch-free.
// =====================================================================
template<int ACT>
__global__ __launch_bounds__(256)
void gemm_k(const __bf16* __restrict__ A, const __bf16* __restrict__ B,
            const float* __restrict__ bias, const float* __restrict__ res,
            float* __restrict__ Cf, __bf16* __restrict__ Cb,
            int M, int N, int K, int lda, int ldb, int ldc,
            long long sA, long long sB, long long sC, float scale)
{
    __shared__ __align__(16) __bf16 As[128][40];
    __shared__ __align__(16) __bf16 Bs[64][40];

    long long z = blockIdx.z;
    A += z * sA;
    B += z * sB;
    if (Cf)  Cf  += z * sC;
    if (Cb)  Cb  += z * sC;
    if (res) res += z * sC;

    const int m0  = blockIdx.y * 128;
    const int n0  = blockIdx.x * 64;
    const int tid = threadIdx.x;
    const int lane = tid & 31;
    const int wid  = tid >> 5;
    const int wm = (wid & 3) * 32;   // wave row offset in block tile
    const int wn = (wid >> 2) * 32;  // wave col offset

    // staging descriptors (hoisted; row-clamped so loads are always valid)
    const int a_row = tid >> 1;
    const int a_ch  = (tid & 1) * 16;
    const int a_gm  = m0 + a_row;
    const __bf16* a_ptr = A + (size_t)(a_gm < M ? a_gm : (M - 1)) * lda + a_ch;

    const int b_n  = tid & 63;
    const int b_kb = (tid >> 6) * 8;
    const int b_gn = n0 + b_n;
    const __bf16* b_ptr = B + (size_t)(b_gn < N ? b_gn : (N - 1)) * ldb + b_kb;

    FragC acc[2][2];
#pragma unroll
    for (int i = 0; i < 2; ++i)
#pragma unroll
        for (int j = 0; j < 2; ++j)
#pragma unroll
            for (int r = 0; r < 8; ++r) acc[i][j].f[r] = 0.0f;

    for (int k0 = 0; k0 < K; k0 += 32) {
        // ---- branch-free vector staging: 2x16B (A) + 1x16B (B) ----
        const uint4 av0 = *(const uint4*)(a_ptr + k0);
        const uint4 av1 = *(const uint4*)(a_ptr + k0 + 8);
        const uint4 bv  = *(const uint4*)(b_ptr + k0);
        *(uint4*)&As[a_row][a_ch]     = av0;
        *(uint4*)&As[a_row][a_ch + 8] = av1;
        *(uint4*)&Bs[b_n][b_kb]       = bv;
        __syncthreads();

        // ---- fragments per CDNA5 wave32 WMMA layout ----
        const int mr = lane & 15;
        const int kh = lane >> 4;
        FragAB af[2], bfr[2];
#pragma unroll
        for (int i = 0; i < 2; ++i) {
            const int row = wm + i * 16 + mr;
            af[i].q[0] = *(const uint4*)&As[row][kh * 8];       // K kh*8..+7
            af[i].q[1] = *(const uint4*)&As[row][16 + kh * 8];  // K 16+kh*8..+7
        }
#pragma unroll
        for (int j = 0; j < 2; ++j) {
            const int col = wn + j * 16 + mr;
            bfr[j].q[0] = *(const uint4*)&Bs[col][kh * 8];      // K kh*8..+7
            bfr[j].q[1] = *(const uint4*)&Bs[col][16 + kh * 8]; // K 16+kh*8..+7
        }
#pragma unroll
        for (int i = 0; i < 2; ++i)
#pragma unroll
            for (int j = 0; j < 2; ++j)
                acc[i][j].v = __builtin_amdgcn_wmma_f32_16x16x32_bf16(
                    false, af[i].v, false, bfr[j].v, (short)0, acc[i][j].v, false, false);
        __syncthreads();
    }

    // ---- epilogue ----
#pragma unroll
    for (int i = 0; i < 2; ++i) {
#pragma unroll
        for (int j = 0; j < 2; ++j) {
            const int n = n0 + wn + j * 16 + (lane & 15);
#pragma unroll
            for (int r = 0; r < 8; ++r) {
                const int m = m0 + wm + i * 16 + r + ((lane >> 4) << 3);
                if (m < M && n < N) {
                    float v = acc[i][j].f[r] * scale;
                    if (bias) v += bias[n];
                    if (res)  v += res[(size_t)m * ldc + n];
                    if (ACT == 1) v = v > 0.0f ? v : 0.0f;
                    if (Cf) Cf[(size_t)m * ldc + n] = v;
                    if (Cb) Cb[(size_t)m * ldc + n] = (__bf16)v;
                }
            }
        }
    }
}

// =====================================================================
// Elementwise / reduction kernels
// =====================================================================
__global__ void k_zero(float* p, int n) {
    int i = blockIdx.x * blockDim.x + threadIdx.x;
    if (i < n) p[i] = 0.0f;
}

// weight convert f32 [K,N] -> bf16 [N,Kp] (transposed, K-pad zeroed)
__global__ void k_f2bt(const float* __restrict__ s, __bf16* __restrict__ d,
                       int K, int N, int Kp) {
    long long i = (long long)blockIdx.x * blockDim.x + threadIdx.x;
    if (i >= (long long)N * Kp) return;
    int n = (int)(i / Kp); int k = (int)(i % Kp);
    d[i] = (k < K) ? (__bf16)s[(size_t)k * N + n] : (__bf16)0.0f;
}

// A_dil[row][c<144] = data_in[row][c/4], pad zero  (bf16, stride 160)
__global__ void k_dil(const float* __restrict__ din, __bf16* __restrict__ A) {
    long long i = (long long)blockIdx.x * blockDim.x + threadIdx.x;
    if (i >= TBTOK * DMODP) return;
    long long row = i / DMODP; int c = (int)(i % DMODP);
    A[i] = (c < DMOD) ? (__bf16)din[row * VV + (c >> 2)] : (__bf16)0.0f;
}

// X[(b*36+v)][t*4+d] = h[(t*256+b)*144 + v*4+d]  (bf16, stride 864, pad 0)
__global__ void k_resh(const float* __restrict__ h, __bf16* __restrict__ X) {
    long long i = (long long)blockIdx.x * blockDim.x + threadIdx.x;
    if (i >= 9216LL * INCHP) return;
    int f = (int)(i % INCHP); int node = (int)(i / INCHP);
    if (f >= INCH) { X[i] = (__bf16)0.0f; return; }
    int b = node / VV, v = node % VV;
    int t = f >> 2, d = f & 3;
    X[i] = (__bf16)h[((size_t)t * BB + b) * DMOD + v * 4 + d];
}

// s1[i]=H[i]·a[0:860], s2[i]=H[i]·a[860:1720]
__global__ void k_gscore(const float* __restrict__ H, const float* __restrict__ a,
                         float* __restrict__ s1, float* __restrict__ s2) {
    int node = blockIdx.x; int tid = threadIdx.x;
    const float* hr = H + (size_t)node * INCH;
    float p1 = 0.f, p2 = 0.f;
    for (int f = tid; f < INCH; f += 256) { float hv = hr[f]; p1 += hv * a[f]; p2 += hv * a[INCH + f]; }
    __shared__ float r1[256], r2[256];
    r1[tid] = p1; r2[tid] = p2; __syncthreads();
    for (int s = 128; s > 0; s >>= 1) { if (tid < s) { r1[tid] += r1[tid + s]; r2[tid] += r2[tid + s]; } __syncthreads(); }
    if (tid == 0) { s1[node] = r1[0]; s2[node] = r2[0]; }
}

// per (b,dst): softmax over src of leaky_relu(s1[src]+s2[dst])*ew
__global__ void k_gattn(const float* __restrict__ s1, const float* __restrict__ s2,
                        const float* __restrict__ ew, float* __restrict__ alpha) {
    int i = blockIdx.x * blockDim.x + threadIdx.x;
    if (i >= 9216) return;
    int b = i / VV, dst = i % VV;
    const float s2i = s2[i];
    float mx = -3.0e38f;
    for (int s = 0; s < VV; ++s) {
        float v = s1[b * VV + s] + s2i;
        v = v > 0.f ? v : 0.2f * v;
        if (ew) v *= ew[(size_t)b * EE + s * VV + dst];
        mx = fmaxf(mx, v);
    }
    float sum = 0.f;
    for (int s = 0; s < VV; ++s) {
        float v = s1[b * VV + s] + s2i;
        v = v > 0.f ? v : 0.2f * v;
        if (ew) v *= ew[(size_t)b * EE + s * VV + dst];
        sum += expf(v - mx);
    }
    const float inv = 1.0f / sum;
    for (int s = 0; s < VV; ++s) {
        float v = s1[b * VV + s] + s2i;
        v = v > 0.f ? v : 0.2f * v;
        if (ew) v *= ew[(size_t)b * EE + s * VV + dst];
        alpha[(size_t)b * EE + s * VV + dst] = expf(v - mx) * inv;
    }
}

// G[dst] = sum_src alpha[b,src,dst] * H[b,src]
// Gf (f32, stride 860) and/or Gb (bf16, stride 864, pad zeroed)
__global__ void k_gagg(const float* __restrict__ H, const float* __restrict__ alpha,
                       float* __restrict__ Gf, __bf16* __restrict__ Gb) {
    int node = blockIdx.y;
    int f = blockIdx.x * 256 + threadIdx.x;
    if (f >= INCHP) return;
    int b = node / VV, dst = node % VV;
    float s = 0.f;
    if (f < INCH) {
        for (int src = 0; src < VV; ++src) {
            float al = alpha[(size_t)b * EE + src * VV + dst];
            s += al * H[((size_t)b * VV + src) * INCH + f];
        }
    }
    if (Gf && f < INCH) Gf[(size_t)node * INCH + f] = s;
    if (Gb) Gb[(size_t)node * INCHP + f] = (__bf16)s;
}

// mean pairwise L2 between alpha vectors (E-dim) over B*B pairs
__global__ void k_dist(const float* __restrict__ alpha, float* __restrict__ out) {
    int pair = blockIdx.x;
    int i = pair >> 8, j = pair & 255;
    int tid = threadIdx.x;
    const float* ai = alpha + (size_t)i * EE;
    const float* aj = alpha + (size_t)j * EE;
    float s = 0.f;
    for (int e = tid; e < EE; e += 256) { float d = ai[e] - aj[e]; s += d * d; }
    __shared__ float red[256];
    red[tid] = s; __syncthreads();
    for (int st = 128; st > 0; st >>= 1) { if (tid < st) red[tid] += red[tid + st]; __syncthreads(); }
    if (tid == 0) atomicAdd(out, sqrtf(fmaxf(red[0], 0.f)) * (1.0f / 65536.0f));
}

// pe[t,b,k<8]=sin(t/ts[k]); pe[t,b,8+k]=cos; ts[k]=215^(k/7)
__global__ void k_pe(const float* __restrict__ times, float* __restrict__ pe) {
    int i = blockIdx.x * blockDim.x + threadIdx.x;
    if (i >= (int)TBTOK) return;
    float t = times[i];
    float* o = pe + (size_t)i * PED;
#pragma unroll
    for (int k = 0; k < 8; ++k) {
        float ts = powf(215.0f, (float)k * (1.0f / 7.0f));
        float sc = t / ts;
        o[k] = sinf(sc); o[8 + k] = cosf(sc);
    }
}

// x[t,b, v*20 + (0..3 gat | 4..19 pe)]; xb row stride 736, pad zeroed
__global__ void k_build(const float* __restrict__ G, const float* __restrict__ pe,
                        float* __restrict__ x, __bf16* __restrict__ xb) {
    int i = blockIdx.x * blockDim.x + threadIdx.x;
    if (i >= (int)(TBTOK * VV)) return;
    int v = i % VV; long long tb = i / VV;
    int b = (int)(tb % BB); int t = (int)(tb / BB);
    size_t basef = (size_t)tb * DTR + v * 20;
    size_t baseb = (size_t)tb * DTRP + v * 20;
    const float* g = G + ((size_t)b * VV + v) * INCH + t * 4;
#pragma unroll
    for (int d = 0; d < 4; ++d) { float f = g[d]; x[basef + d] = f; xb[baseb + d] = (__bf16)f; }
    const float* p = pe + (size_t)tb * PED;
#pragma unroll
    for (int k = 0; k < PED; ++k) { float f = p[k]; x[basef + 4 + k] = f; xb[baseb + 4 + k] = (__bf16)f; }
    if (v == 0) {
#pragma unroll
        for (int c = DTR; c < DTRP; ++c) xb[(size_t)tb * DTRP + c] = (__bf16)0.0f;
    }
}

// [T,B,720] (stride 720) -> [B*H, T, 192]  (K-pad zeroed; dword copies)
__global__ void k_p2h(const __bf16* __restrict__ in, __bf16* __restrict__ out) {
    long long i = (long long)blockIdx.x * blockDim.x + threadIdx.x;
    if (i >= (long long)BH * TT * (HDP / 2)) return;
    int d2 = (int)(i % (HDP / 2)); long long r = i / (HDP / 2);
    int t = (int)(r % TT); int bh = (int)(r / TT);
    int b = bh >> 2, h = bh & 3;
    unsigned int val = 0;
    if (d2 < HDIM / 2)
        val = *(const unsigned int*)(in + ((size_t)t * BB + b) * DTR + h * HDIM + d2 * 2);
    *(unsigned int*)(out + ((size_t)bh * TT + t) * HDP + d2 * 2) = val;
}

// [T,B,720] -> [B*H, 180, 224]  (transposed heads, K-pad zeroed)
__global__ void k_p2hT(const __bf16* __restrict__ in, __bf16* __restrict__ out) {
    long long i = (long long)blockIdx.x * blockDim.x + threadIdx.x;
    if (i >= (long long)BH * HDIM * TTP) return;
    int t = (int)(i % TTP); long long r = i / TTP;
    int d = (int)(r % HDIM); int bh = (int)(r / HDIM);
    int b = bh >> 2, h = bh & 3;
    out[i] = (t < TT) ? in[((size_t)t * BB + b) * DTR + h * HDIM + d] : (__bf16)0.0f;
}

// [B*H, T, 192] -> [T,B,736-strided 720 cols]
__global__ void k_h2p(const __bf16* __restrict__ in, __bf16* __restrict__ out) {
    long long i = (long long)blockIdx.x * blockDim.x + threadIdx.x;
    if (i >= (long long)BH * TT * (HDIM / 2)) return;
    int d2 = (int)(i % (HDIM / 2)); long long r = i / (HDIM / 2);
    int t = (int)(r % TT); int bh = (int)(r / TT);
    int b = bh >> 2, h = bh & 3;
    const unsigned int* src = (const unsigned int*)(in + ((size_t)bh * TT + t) * HDP + d2 * 2);
    unsigned int* dst = (unsigned int*)(out + ((size_t)t * BB + b) * DTRP + h * HDIM + d2 * 2);
    *dst = *src;
}

// masked softmax over keys; scores f32 (stride 215) -> attn bf16 (stride 224, pad 0)
__global__ void k_softmax(const float* __restrict__ sc, const int* __restrict__ lengths,
                          __bf16* __restrict__ at) {
    int row = blockIdx.x;            // bh*T + q
    int bh = row / TT; int b = bh >> 2;
    int tid = threadIdx.x;
    const float* s = sc + (size_t)row * TT;
    int len = lengths[b];
    float v = -3.0e38f;
    if (tid < TT) v = (tid < len) ? s[tid] : -1.0e9f;
    __shared__ float red[256];
    red[tid] = v; __syncthreads();
    for (int st = 128; st > 0; st >>= 1) { if (tid < st) red[tid] = fmaxf(red[tid], red[tid + st]); __syncthreads(); }
    float mx = red[0]; __syncthreads();
    float p = 0.f;
    if (tid < TT) p = expf(v - mx);
    red[tid] = p; __syncthreads();
    for (int st = 128; st > 0; st >>= 1) { if (tid < st) red[tid] += red[tid + st]; __syncthreads(); }
    float inv = 1.0f / red[0];
    if (tid < TTP) at[(size_t)row * TTP + tid] = (__bf16)((tid < TT) ? p * inv : 0.0f);
}

// layernorm over 720, in-place f32 + bf16 shadow (stride 736, pad zeroed)
__global__ void k_ln(float* __restrict__ x, __bf16* __restrict__ xb,
                     const float* __restrict__ g, const float* __restrict__ bta) {
    int row = blockIdx.x; int tid = threadIdx.x;
    float* xr = x + (size_t)row * DTR;
    float vals[3]; float s = 0.f, s2 = 0.f;
#pragma unroll
    for (int u = 0; u < 3; ++u) {
        int c = tid + u * 256;
        float f = (c < DTR) ? xr[c] : 0.f;
        vals[u] = f; s += f; s2 += f * f;
    }
    __shared__ float r1[256], r2[256];
    r1[tid] = s; r2[tid] = s2; __syncthreads();
    for (int st = 128; st > 0; st >>= 1) { if (tid < st) { r1[tid] += r1[tid + st]; r2[tid] += r2[tid + st]; } __syncthreads(); }
    float mu = r1[0] * (1.0f / DTR);
    float var = r2[0] * (1.0f / DTR) - mu * mu;
    float inv = rsqrtf(var + 1e-5f);
#pragma unroll
    for (int u = 0; u < 3; ++u) {
        int c = tid + u * 256;
        if (c < DTR) {
            float y = (vals[u] - mu) * inv * g[c] + bta[c];
            xr[c] = y; xb[(size_t)row * DTRP + c] = (__bf16)y;
        } else if (c < DTRP) {
            xb[(size_t)row * DTRP + c] = (__bf16)0.0f;
        }
    }
}

// out[b, v*20+c] = sum_t x*(1-mask) / (sum_t mask + 1)
__global__ void k_agg(const float* __restrict__ x, const float* __restrict__ mask,
                      float* __restrict__ out) {
    int i = blockIdx.x * blockDim.x + threadIdx.x;
    if (i >= BB * DTR) return;
    int b = i / DTR; int rem = i % DTR; int v = rem / 20;
    float s = 0.f, cnt = 0.f;
    for (int t = 0; t < TT; ++t) {
        size_t tb = (size_t)t * BB + b;
        float m = mask[tb * VV + v];
        cnt += m;
        s += x[tb * DTR + rem] * (1.0f - m);
    }
    out[i] = s / (cnt + 1.0f);
}

// =====================================================================
// host orchestration
// =====================================================================
static inline dim3 g1(long long n) { return dim3((unsigned)((n + 255) / 256)); }

extern "C" void kernel_launch(void* const* d_in, const int* in_sizes, int n_in,
                              void* d_out, int out_size, void* d_ws, size_t ws_size,
                              hipStream_t stream) {
    (void)in_sizes; (void)n_in; (void)out_size;
    const float* data_in  = (const float*)d_in[0];
    const float* times_in = (const float*)d_in[1];
    const float* mask_in  = (const float*)d_in[2];
    const int*   lengths  = (const int*)  d_in[3];
    const float* W_dil    = (const float*)d_in[4];
    const float* b_dil    = (const float*)d_in[5];
    const float* W_gat1   = (const float*)d_in[6];
    const float* a_gat1   = (const float*)d_in[7];
    const float* W_gat2   = (const float*)d_in[8];
    const float* a_gat2   = (const float*)d_in[9];
    float* outp = (float*)d_out;

    unsigned char* ws = (unsigned char*)d_ws;
    size_t o = 0;
    auto alloc = [&](size_t bytes) { size_t cur = o; o += (bytes + 255) & ~(size_t)255; return cur; };

    size_t oWDIL = alloc((size_t)DMOD * DMODP * 2);
    size_t oWG1  = alloc((size_t)INCH * INCHP * 2);
    size_t oWG2  = alloc((size_t)INCH * INCHP * 2);
    size_t oWQ[2], oWK[2], oWV[2], oWO[2], oW1[2], oW2[2];
    for (int l = 0; l < 2; ++l) {
        oWQ[l] = alloc((size_t)DTR * DTRP * 2);
        oWK[l] = alloc((size_t)DTR * DTRP * 2);
        oWV[l] = alloc((size_t)DTR * DTRP * 2);
        oWO[l] = alloc((size_t)DTR * DTRP * 2);
        oW1[l] = alloc((size_t)NHIDD * DTRP * 2);   // W1^T [512,736]
        oW2[l] = alloc((size_t)DTR * NHIDD * 2);    // W2^T [720,512]
    }
    size_t oX  = alloc((size_t)TBTOK * DTR * 4);       // x f32 (stride 720)
    size_t oXB = alloc((size_t)TBTOK * DTRP * 2);      // x bf16 (stride 736)
    size_t oQH = alloc((size_t)BH * TT * HDP * 2);     // Q heads [bh,215,192] / ctx heads
    size_t oKH = alloc((size_t)BH * TT * HDP * 2);     // K heads / ctx merged [55040,736]
    size_t oVH = alloc((size_t)BH * HDIM * TTP * 2);   // V^T heads [bh,180,224] / ff bf16
    size_t oSC = alloc((size_t)BH * TT * TT * 4);      // scores f32 / early scratch / TMP
    size_t oAT = alloc((size_t)BH * TT * TTP * 2);     // attn bf16 (stride 224)
    if (ws_size < o) return;

    // early-phase aliases inside the (not-yet-used) scores region (189.3MB)
    size_t oH    = oSC;                         // h f32 [55040,144]       31.7MB
    size_t oADIL = oSC + (size_t)32  * 1048576; // dilated A bf16 [.,160]  17.6MB
    size_t oXN   = oSC + (size_t)50  * 1048576; // node feats bf16 [.,864] 15.9MB
    size_t oHN   = oSC + (size_t)66  * 1048576; // H f32 [9216,860]        31.7MB
    size_t oGN   = oSC + (size_t)98  * 1048576; // G f32 [9216,860]        31.7MB
    size_t oA1   = oSC + (size_t)130 * 1048576; // alpha1                  1.33MB
    size_t oA2   = oSC + (size_t)132 * 1048576; // alpha2                  1.33MB
    size_t oS1   = oSC + (size_t)134 * 1048576;
    size_t oS2   = oS1 + 40960;
    size_t oPEb  = oSC + (size_t)135 * 1048576; // pos enc f32             3.5MB
    size_t oTMP  = oSC;                         // pre-permute QKV bf16 (stride 720)

    auto B16 = [&](size_t off) { return (__bf16*)(ws + off); };
    auto F32 = [&](size_t off) { return (float*)(ws + off); };

    // ---- 0. init distance accumulator ----
    k_zero<<<1, 256, 0, stream>>>(outp + BB * DTR, 1);

    // ---- 1. weights -> bf16, transposed to [N,Kp], pad zeroed ----
    k_f2bt<<<g1((long long)DMOD * DMODP), 256, 0, stream>>>(W_dil, B16(oWDIL), DMOD, DMOD, DMODP);
    k_f2bt<<<g1((long long)INCH * INCHP), 256, 0, stream>>>(W_gat1, B16(oWG1), INCH, INCH, INCHP);
    k_f2bt<<<g1((long long)INCH * INCHP), 256, 0, stream>>>(W_gat2, B16(oWG2), INCH, INCH, INCHP);
    for (int l = 0; l < 2; ++l) {
        int base = 10 + l * 16;
        k_f2bt<<<g1((long long)DTR * DTRP), 256, 0, stream>>>((const float*)d_in[base + 0], B16(oWQ[l]), DTR, DTR, DTRP);
        k_f2bt<<<g1((long long)DTR * DTRP), 256, 0, stream>>>((const float*)d_in[base + 1], B16(oWK[l]), DTR, DTR, DTRP);
        k_f2bt<<<g1((long long)DTR * DTRP), 256, 0, stream>>>((const float*)d_in[base + 2], B16(oWV[l]), DTR, DTR, DTRP);
        k_f2bt<<<g1((long long)DTR * DTRP), 256, 0, stream>>>((const float*)d_in[base + 3], B16(oWO[l]), DTR, DTR, DTRP);
        k_f2bt<<<g1((long long)NHIDD * DTRP), 256, 0, stream>>>((const float*)d_in[base + 8], B16(oW1[l]), DTR, NHIDD, DTRP);
        k_f2bt<<<g1((long long)DTR * NHIDD), 256, 0, stream>>>((const float*)d_in[base + 10], B16(oW2[l]), NHIDD, DTR, NHIDD);
    }

    // ---- 2. dilated embed: h = relu(repeat(data)@Wdil + b) ----
    k_dil<<<g1(TBTOK * DMODP), 256, 0, stream>>>(data_in, B16(oADIL));
    {
        dim3 grid((DMOD + 63) / 64, (unsigned)((TBTOK + 127) / 128), 1);
        gemm_k<1><<<grid, 256, 0, stream>>>(B16(oADIL), B16(oWDIL), b_dil, nullptr,
            F32(oH), nullptr, (int)TBTOK, DMOD, DMODP, DMODP, DMODP, DMOD, 0, 0, 0, 1.0f);
    }

    // ---- 3. GAT layer 1 ----
    k_resh<<<g1(9216LL * INCHP), 256, 0, stream>>>(F32(oH), B16(oXN));
    {
        dim3 grid((INCH + 63) / 64, (9216 + 127) / 128, 1);
        gemm_k<0><<<grid, 256, 0, stream>>>(B16(oXN), B16(oWG1), nullptr, nullptr,
            F32(oHN), nullptr, 9216, INCH, INCHP, INCHP, INCHP, INCH, 0, 0, 0, 1.0f);
    }
    k_gscore<<<9216, 256, 0, stream>>>(F32(oHN), a_gat1, F32(oS1), F32(oS2));
    k_gattn<<<g1(9216), 256, 0, stream>>>(F32(oS1), F32(oS2), nullptr, F32(oA1));
    { dim3 grid((INCHP + 255) / 256, 9216); k_gagg<<<grid, 256, 0, stream>>>(F32(oHN), F32(oA1), nullptr, B16(oXN)); }

    // ---- 4. GAT layer 2 (edge weights = alpha1) ----
    {
        dim3 grid((INCH + 63) / 64, (9216 + 127) / 128, 1);
        gemm_k<0><<<grid, 256, 0, stream>>>(B16(oXN), B16(oWG2), nullptr, nullptr,
            F32(oHN), nullptr, 9216, INCH, INCHP, INCHP, INCHP, INCH, 0, 0, 0, 1.0f);
    }
    k_gscore<<<9216, 256, 0, stream>>>(F32(oHN), a_gat2, F32(oS1), F32(oS2));
    k_gattn<<<g1(9216), 256, 0, stream>>>(F32(oS1), F32(oS2), F32(oA1), F32(oA2));
    { dim3 grid((INCHP + 255) / 256, 9216); k_gagg<<<grid, 256, 0, stream>>>(F32(oHN), F32(oA2), F32(oGN), nullptr); }

    // ---- 5. alpha distance ----
    k_dist<<<65536, 256, 0, stream>>>(F32(oA2), outp + BB * DTR);

    // ---- 6. assemble transformer input ----
    k_pe<<<g1(TBTOK), 256, 0, stream>>>(times_in, F32(oPEb));
    k_build<<<g1(TBTOK * VV), 256, 0, stream>>>(F32(oGN), F32(oPEb), F32(oX), B16(oXB));

    // ---- 7. transformer layers ----
    const float iscale = 0.07453559925f; // 1/sqrt(180)
    const long long SQK = (long long)TT * HDP;    // 41280 (Q/K head batch stride)
    const long long SVT = (long long)HDIM * TTP;  // 40320 (V^T head batch stride)
    const long long SSC = (long long)TT * TT;     // 46225
    const long long SAT = (long long)TT * TTP;    // 48160
    for (int l = 0; l < 2; ++l) {
        int base = 10 + l * 16;
        const float* bq = (const float*)d_in[base + 4];
        const float* bk = (const float*)d_in[base + 5];
        const float* bv = (const float*)d_in[base + 6];
        const float* bo = (const float*)d_in[base + 7];
        const float* b1 = (const float*)d_in[base + 9];
        const float* b2 = (const float*)d_in[base + 11];
        const float* ln1g = (const float*)d_in[base + 12];
        const float* ln1b = (const float*)d_in[base + 13];
        const float* ln2g = (const float*)d_in[base + 14];
        const float* ln2b = (const float*)d_in[base + 15];

        dim3 gproj((DTR + 63) / 64, (unsigned)((TBTOK + 127) / 128), 1);
        long long nP2H = (long long)BH * TT * (HDP / 2);
        long long nH2P = (long long)BH * TT * (HDIM / 2);

        // Q, K, V projections -> head layouts (V transposed for ctx GEMM)
        gemm_k<0><<<gproj, 256, 0, stream>>>(B16(oXB), B16(oWQ[l]), bq, nullptr,
            nullptr, B16(oTMP), (int)TBTOK, DTR, DTRP, DTRP, DTRP, DTR, 0, 0, 0, 1.0f);
        k_p2h<<<g1(nP2H), 256, 0, stream>>>(B16(oTMP), B16(oQH));
        gemm_k<0><<<gproj, 256, 0, stream>>>(B16(oXB), B16(oWK[l]), bk, nullptr,
            nullptr, B16(oTMP), (int)TBTOK, DTR, DTRP, DTRP, DTRP, DTR, 0, 0, 0, 1.0f);
        k_p2h<<<g1(nP2H), 256, 0, stream>>>(B16(oTMP), B16(oKH));
        gemm_k<0><<<gproj, 256, 0, stream>>>(B16(oXB), B16(oWV[l]), bv, nullptr,
            nullptr, B16(oTMP), (int)TBTOK, DTR, DTRP, DTRP, DTRP, DTR, 0, 0, 0, 1.0f);
        k_p2hT<<<g1((long long)BH * HDIM * TTP), 256, 0, stream>>>(B16(oTMP), B16(oVH));

        // scores = scale * Q @ K^T   (batched over B*H; K=192 padded head dim)
        {
            dim3 grid((TT + 63) / 64, (TT + 127) / 128, BH);
            gemm_k<0><<<grid, 256, 0, stream>>>(B16(oQH), B16(oKH), nullptr, nullptr,
                F32(oSC), nullptr, TT, TT, HDP, HDP, HDP, TT, SQK, SQK, SSC, iscale);
        }
        // masked softmax -> attn bf16 (stride 224, pad zero)
        k_softmax<<<BH * TT, 256, 0, stream>>>(F32(oSC), lengths, B16(oAT));
        // ctx = attn @ V : A=attn [215,224], B=V^T [180,224]  (ctx heads -> QH)
        {
            dim3 grid((HDIM + 63) / 64, (TT + 127) / 128, BH);
            gemm_k<0><<<grid, 256, 0, stream>>>(B16(oAT), B16(oVH), nullptr, nullptr,
                nullptr, B16(oQH), TT, HDIM, TTP, TTP, TTP, HDP, SAT, SVT, SQK, 1.0f);
        }
        // merge heads -> [T,B,736-strided] bf16 (reuse KH)
        k_h2p<<<g1(nH2P), 256, 0, stream>>>(B16(oQH), B16(oKH));
        // x = x + ctx @ Wo + bo ; LN1
        gemm_k<0><<<gproj, 256, 0, stream>>>(B16(oKH), B16(oWO[l]), bo, F32(oX),
            F32(oX), nullptr, (int)TBTOK, DTR, DTRP, DTRP, DTRP, DTR, 0, 0, 0, 1.0f);
        k_ln<<<(unsigned)TBTOK, 256, 0, stream>>>(F32(oX), B16(oXB), ln1g, ln1b);

        // ff = relu(x@W1+b1) ; x = LN2(x + ff@W2 + b2)
        {
            dim3 grid((NHIDD + 63) / 64, (unsigned)((TBTOK + 127) / 128), 1);
            gemm_k<1><<<grid, 256, 0, stream>>>(B16(oXB), B16(oW1[l]), b1, nullptr,
                nullptr, B16(oVH), (int)TBTOK, NHIDD, DTRP, DTRP, DTRP, NHIDD, 0, 0, 0, 1.0f);
        }
        gemm_k<0><<<gproj, 256, 0, stream>>>(B16(oVH), B16(oW2[l]), b2, F32(oX),
            F32(oX), nullptr, (int)TBTOK, DTR, NHIDD, NHIDD, NHIDD, DTR, 0, 0, 0, 1.0f);
        k_ln<<<(unsigned)TBTOK, 256, 0, stream>>>(F32(oX), B16(oXB), ln2g, ln2b);
    }

    // ---- 8. masked time aggregation -> d_out[0 .. 184319] ----
    k_agg<<<g1(BB * DTR), 256, 0, stream>>>(F32(oX), mask_in, outp);
}
